// SimpleMPNN_11940009083287
// MI455X (gfx1250) — compile-verified, matrix operand order checked by reading, not run on hardware
//
#include <hip/hip_runtime.h>
#include <hip/hip_bf16.h>
#include <stdint.h>

// ---------------------------------------------------------------------------
// MPNN for MI455X (gfx1250, wave32, WMMA).
//  - 5 linear layers as bf16 WMMA GEMMs (f32 accumulate), weights + A-tile in LDS
//  - float4 staging loads, packed 4x bf16 LDS stores
//  - block-level full/partial split: hot path stores are branch-free
//  - branch-free fast tanh (v_exp_f32 + v_rcp_f32), saturates correctly
//  - edge scatter-add via global f32 atomics, grid-stride waves + prefetch
// ---------------------------------------------------------------------------

typedef __attribute__((ext_vector_type(16))) __bf16        v16bf;
typedef __attribute__((ext_vector_type(4)))  __bf16        v4bf;
typedef __attribute__((ext_vector_type(8)))  float         v8f;
typedef __attribute__((ext_vector_type(8)))  unsigned int  v8u;

// K-offset pattern for the 16-bit 16x32 A/B fragment (ISA 7.12.2):
// lanes 0-15 hold K pairs {0..7,16..23}, lanes 16-31 hold {8..15,24..31}.
__device__ __forceinline__ int kOff(int v, int half) {
  return ((v & 4) << 2) | (half << 3) | ((v & 3) << 1);
}

// tanh(x) = 1 - 2/(exp(2x)+1); exp overflow/underflow yields exact +/-1.
__device__ __forceinline__ float fast_tanh(float x) {
  float e = __expf(2.0f * x);
  return 1.0f - 2.0f * __builtin_amdgcn_rcpf(e + 1.0f);
}

__device__ __forceinline__ v4bf pack4(float4 v) {
  v4bf r = {(__bf16)v.x, (__bf16)v.y, (__bf16)v.z, (__bf16)v.w};
  return r;
}

// out[M,N] = act(A[M,128] @ W[N,128]^T + bias[N]); act = tanh if ACT.
// Block = 128 threads (4 waves), each block computes a 64-row stripe.
template <int N, bool ACT>
__global__ __launch_bounds__(128) void gemm_bias_act(
    const float* __restrict__ A, const float* __restrict__ W,
    const float* __restrict__ bias, float* __restrict__ out, int M) {
  __shared__ __bf16 ldsA[64 * 128];
  __shared__ __bf16 ldsB[N * 128];
  __shared__ float  ldsBias[N];

  const int tid = threadIdx.x;
  const int m0  = blockIdx.x * 64;
  const bool fullBlock = (m0 + 64 <= M);

  // Stage 64x128 activation tile (f32 -> bf16), float4 loads, zero-pad past M.
  const float4 fz = {0.f, 0.f, 0.f, 0.f};
  for (int i4 = tid; i4 < 64 * 32; i4 += 128) {
    int rr = i4 >> 5, k4 = i4 & 31;
    int gr = m0 + rr;
    float4 v = (gr < M) ? ((const float4*)A)[(size_t)gr * 32 + k4] : fz;
    *(v4bf*)&ldsA[i4 * 4] = pack4(v);
  }
  // Stage weights (N x 128, row-major) as bf16, float4 loads.
  for (int i4 = tid; i4 < N * 32; i4 += 128)
    *(v4bf*)&ldsB[i4 * 4] = pack4(((const float4*)W)[i4]);
  if (tid < N) ldsBias[tid] = bias[tid];
  __syncthreads();

  const int lane = tid & 31;
  const int wv   = tid >> 5;   // wave id: rows [wv*16, wv*16+16)
  const int half = lane >> 4;
  const int mr   = lane & 15;
  const int rowA = wv * 16 + mr;

  // Preload all 4 A fragments (K = 0,32,64,96).
  v16bf afrag[4];
#pragma unroll
  for (int kt = 0; kt < 4; ++kt) {
    v8u raw;
#pragma unroll
    for (int v = 0; v < 8; ++v)
      raw[v] = *(const unsigned int*)&ldsA[rowA * 128 + kt * 32 + kOff(v, half)];
    afrag[kt] = __builtin_bit_cast(v16bf, raw);
  }

#pragma unroll
  for (int nt = 0; nt < N / 16; ++nt) {
    float bb = ldsBias[nt * 16 + mr];
    v8f acc = {bb, bb, bb, bb, bb, bb, bb, bb};
#pragma unroll
    for (int kt = 0; kt < 4; ++kt) {
      // B column n = row n of W: same fragment pattern as A.
      v8u raw;
#pragma unroll
      for (int v = 0; v < 8; ++v)
        raw[v] = *(const unsigned int*)&ldsB[(nt * 16 + mr) * 128 + kt * 32 + kOff(v, half)];
      v16bf bfrag = __builtin_bit_cast(v16bf, raw);
      acc = __builtin_amdgcn_wmma_f32_16x16x32_bf16(
          false, afrag[kt], false, bfrag, (short)0, acc, false, false);
    }
    // D layout: lane%16 = column, reg r -> row 8*(lane/16)+r.
    const int mbase = m0 + wv * 16 + 8 * half;
    float* op = out + (size_t)mbase * N + nt * 16 + mr;
    if (fullBlock) {
      // Hot path (781/782 blocks): branch-free, base + immediate offsets.
#pragma unroll
      for (int r = 0; r < 8; ++r) {
        float val = acc[r];
        if (ACT) val = fast_tanh(val);
        op[(size_t)r * N] = val;
      }
    } else {
#pragma unroll
      for (int r = 0; r < 8; ++r) {
        if (mbase + r < M) {
          float val = acc[r];
          if (ACT) val = fast_tanh(val);
          op[(size_t)r * N] = val;
        }
      }
    }
  }
}

__global__ __launch_bounds__(256) void zero_kernel(float4* __restrict__ p, size_t n4) {
  size_t i = (size_t)blockIdx.x * blockDim.x + threadIdx.x;
  size_t stride = (size_t)gridDim.x * blockDim.x;
  const float4 z = {0.0f, 0.0f, 0.0f, 0.0f};
  for (; i < n4; i += stride) p[i] = z;
}

// aggr[row[e], :] += h[col[e], :].  One wave per edge iteration:
// 32 lanes x float4 = 128 floats; prefetch next edge's source row.
__global__ __launch_bounds__(256) void scatter_add_kernel(
    const float* __restrict__ h, const long long* __restrict__ row,
    const long long* __restrict__ col, float* __restrict__ aggr, int E) {
  const int waveStride = gridDim.x * 8;
  const int lane = threadIdx.x & 31;
  int e = blockIdx.x * 8 + (threadIdx.x >> 5);
  for (; e < E; e += waveStride) {
    long long r = row[e];
    long long c = col[e];
    int en = e + waveStride;
    if (en < E) {
      long long cn = col[en];
      __builtin_prefetch(h + cn * 128 + lane * 4, 0, 3);  // global_prefetch
    }
    float4 v = ((const float4*)(h + c * 128))[lane];
    float* dst = aggr + r * 128 + lane * 4;
    atomicAdd(dst + 0, v.x);
    atomicAdd(dst + 1, v.y);
    atomicAdd(dst + 2, v.z);
    atomicAdd(dst + 3, v.w);
  }
}

extern "C" void kernel_launch(void* const* d_in, const int* in_sizes, int n_in,
                              void* d_out, int out_size, void* d_ws, size_t ws_size,
                              hipStream_t stream) {
  const float*     x     = (const float*)d_in[0];
  const long long* edges = (const long long*)d_in[1];
  const float*     W_in  = (const float*)d_in[2];
  const float*     b_in  = (const float*)d_in[3];
  const float*     Wl[3] = {(const float*)d_in[4], (const float*)d_in[6], (const float*)d_in[8]};
  const float*     bl[3] = {(const float*)d_in[5], (const float*)d_in[7], (const float*)d_in[9]};
  const float*     W_out = (const float*)d_in[10];
  const float*     b_out = (const float*)d_in[11];
  float*           out   = (float*)d_out;

  const int M   = 50000;
  const int HID = 128;
  const int E   = in_sizes[1] / 2;  // edge_index is [2, E]
  const long long* rowIdx = edges;      // edge_index[0]
  const long long* colIdx = edges + E;  // edge_index[1]

  float* h    = (float*)d_ws;
  float* aggr = h + (size_t)M * HID;

  const int gemmBlocks = (M + 63) / 64;
  const size_t n4 = (size_t)M * HID / 4;
  const int zeroBlocks = 1024;
  const int scatBlocks = 4096;  // 32768 waves, grid-stride over E edges

  // h = tanh(x @ W_in^T + b_in)
  gemm_bias_act<128, true><<<gemmBlocks, 128, 0, stream>>>(x, W_in, b_in, h, M);

  for (int l = 0; l < 3; ++l) {
    zero_kernel<<<zeroBlocks, 256, 0, stream>>>((float4*)aggr, n4);
    scatter_add_kernel<<<scatBlocks, 256, 0, stream>>>(h, rowIdx, colIdx, aggr, E);
    gemm_bias_act<128, true><<<gemmBlocks, 128, 0, stream>>>(aggr, Wl[l], bl[l], h, M);
  }

  // out = h @ W_out^T + b_out   (no activation)
  gemm_bias_act<64, false><<<gemmBlocks, 128, 0, stream>>>(h, W_out, b_out, out, M);
}
